// Attention_org_46755013984665
// MI455X (gfx1250) — matrix operand order repaired and checked by reading, not compiled, not run on hardware
//
#include <hip/hip_runtime.h>

// ---------------------------------------------------------------------------
// CDNA5 (gfx1250) wave32 WMMA attention pipeline, f16 inputs / f32 accum.
// ---------------------------------------------------------------------------

typedef __attribute__((ext_vector_type(16))) _Float16      v16h;
typedef __attribute__((ext_vector_type(8)))  float         v8f;
typedef __attribute__((ext_vector_type(8)))  unsigned short u16x8;

union FragU { v16h v; u16x8 h[2]; };

__device__ __forceinline__ unsigned short f2h(float x) {
  _Float16 h = (_Float16)x;               // v_cvt_f16_f32, RNE
  return __builtin_bit_cast(unsigned short, h);
}

// A fragment: 16x32 f16, row-major source A[M,K] (lda = K stride).
// Lane L: row = row0 + (L&15); kb = k0 + (L>=16 ? 8 : 0);
// v[0..7] = A[row][kb..kb+7], v[8..15] = A[row][kb+16..kb+23]   (ISA 7.12.2)
__device__ __forceinline__ v16h loadA_frag(const unsigned short* A, int lda,
                                           int row0, int k0, int lane) {
  int r  = row0 + (lane & 15);
  int kb = k0 + ((lane >> 4) << 3);
  const u16x8* p = (const u16x8*)(A + (size_t)r * lda + kb);
  FragU f; f.h[0] = p[0]; f.h[1] = p[2];
  return f.v;
}

// B fragment: 32x16 f16, taken from Bt[N,K] row-major (ldb = K stride).
// Lane L: col = col0 + (L&15); kb = k0 + (L>=16 ? 8 : 0);
// v[0..7] = Bt[col][kb..kb+7], v[8..15] = Bt[col][kb+16..kb+23]
__device__ __forceinline__ v16h loadB_frag(const unsigned short* Bt, int ldb,
                                           int col0, int k0, int lane) {
  int c  = col0 + (lane & 15);
  int kb = k0 + ((lane >> 4) << 3);
  const u16x8* p = (const u16x8*)(Bt + (size_t)c * ldb + kb);
  FragU f; f.h[0] = p[0]; f.h[1] = p[2];
  return f.v;
}

// Generic batched GEMM: D[M,N] = scale * A[M,K] * Bt[N,K]^T
// One wave per block, 64x64 output tile = 4x4 WMMA fragments.
// Batch z -> (b = z/H, h = z%H); per-operand (b,h) strides in elements.
// outF32==0 -> f16 output, outF32==1 -> f32 output (rows < cRowBound only).
__global__ __launch_bounds__(32) void wmma_gemm_kernel(
    const unsigned short* __restrict__ A,
    const unsigned short* __restrict__ Bt,
    void* __restrict__ Cout,
    int K, int lda, int ldb, int ldc,
    long long aSB, long long aSH,
    long long bSB, long long bSH,
    long long cSB, long long cSH,
    int H, int cRowBound, float scale, int outF32)
{
  int z  = blockIdx.z;
  int bb = z / H, hh = z % H;
  A  += (size_t)bb * aSB + (size_t)hh * aSH;
  Bt += (size_t)bb * bSB + (size_t)hh * bSH;
  int m0   = blockIdx.y << 6;
  int n0   = blockIdx.x << 6;
  int lane = threadIdx.x & 31;

  v8f acc[4][4] = {};

  for (int k0 = 0; k0 < K; k0 += 32) {
    v16h af[4];
#pragma unroll
    for (int i = 0; i < 4; ++i) af[i] = loadA_frag(A, lda, m0 + (i << 4), k0, lane);
#pragma unroll
    for (int j = 0; j < 4; ++j) {
      v16h bfrag = loadB_frag(Bt, ldb, n0 + (j << 4), k0, lane);
#pragma unroll
      for (int i = 0; i < 4; ++i) {
        acc[i][j] = __builtin_amdgcn_wmma_f32_16x16x32_f16(
            /*neg_a=*/false, af[i], /*neg_b=*/false, bfrag,
            /*c_mod=*/(short)0, acc[i][j], /*reuse_a=*/false, /*reuse_b=*/false);
      }
    }
  }

  // D layout (f32 16x16): lane L holds col = L&15; VGPR r -> row r + (L>=16)*8.
  int colIn   = lane & 15;
  int rowHalf = (lane >> 4) << 3;
  if (outF32) {
    float* Cf = (float*)Cout + (size_t)bb * cSB + (size_t)hh * cSH;
#pragma unroll
    for (int i = 0; i < 4; ++i)
#pragma unroll
      for (int r = 0; r < 8; ++r) {
        int row = m0 + (i << 4) + rowHalf + r;
        if (row < cRowBound) {
#pragma unroll
          for (int j = 0; j < 4; ++j) {
            int col = n0 + (j << 4) + colIn;
            Cf[(size_t)row * ldc + col] = acc[i][j][r] * scale;
          }
        }
      }
  } else {
    unsigned short* Cb = (unsigned short*)Cout + (size_t)bb * cSB + (size_t)hh * cSH;
#pragma unroll
    for (int i = 0; i < 4; ++i)
#pragma unroll
      for (int r = 0; r < 8; ++r) {
        int row = m0 + (i << 4) + rowHalf + r;
        if (row < cRowBound) {
#pragma unroll
          for (int j = 0; j < 4; ++j) {
            int col = n0 + (j << 4) + colIn;
            Cb[(size_t)row * ldc + col] = f2h(acc[i][j][r] * scale);
          }
        }
      }
  }
}

// fp32 -> f16 with zero row padding: in [batch][rows][cols] -> out [batch][rowsPad][cols]
__global__ __launch_bounds__(256) void convert_pad_kernel(
    const float* __restrict__ in, unsigned short* __restrict__ out,
    int rows, int rowsPad, int cols, long long total)
{
  long long idx = (long long)blockIdx.x * 256 + threadIdx.x;
  if (idx >= total) return;
  int c = (int)(idx % cols);
  long long t = idx / cols;
  int r = (int)(t % rowsPad);
  long long b = t / rowsPad;
  float v = (r < rows) ? in[((size_t)b * rows + r) * cols + c] : 0.0f;
  out[idx] = f2h(v);
}

// Deterministic per-(b,h) sum / sumsq over a [C,KV] score map (one block per map).
__global__ __launch_bounds__(256) void stats_kernel(
    const float* __restrict__ scores, float* __restrict__ stats, int perMap)
{
  int z = blockIdx.x;
  const float* p = scores + (size_t)z * perMap;
  int t = threadIdx.x;
  float s = 0.f, ss = 0.f;
  for (int i = t; i < perMap; i += 256) { float v = p[i]; s += v; ss += v * v; }
  __shared__ float rs[256], rss[256];
  rs[t] = s; rss[t] = ss; __syncthreads();
  for (int k = 128; k > 0; k >>= 1) {
    if (t < k) { rs[t] += rs[t + k]; rss[t] += rss[t + k]; }
    __syncthreads();
  }
  if (t == 0) { stats[2 * z] = rs[0]; stats[2 * z + 1] = rss[0]; }
}

// Instance-norm (per (b,h) map) + row softmax over KV, f32 scores -> f16 probs.
__global__ __launch_bounds__(256) void norm_softmax_kernel(
    const float* __restrict__ scores, unsigned short* __restrict__ probs,
    const float* __restrict__ stats, int Cdim, int KVdim)
{
  int z = blockIdx.y;
  int d = blockIdx.x;
  const float* row = scores + ((size_t)z * Cdim + d) * KVdim;
  unsigned short* orow = probs + ((size_t)z * Cdim + d) * KVdim;
  float cnt  = (float)Cdim * (float)KVdim;
  float m    = stats[2 * z] / cnt;
  float var  = stats[2 * z + 1] / cnt - m * m;
  float rstd = rsqrtf(var + 1e-5f);

  __shared__ float red[256];
  int t = threadIdx.x;
  float x[4];
  float lmax = -3.0e38f;
#pragma unroll
  for (int i = 0; i < 4; ++i) {
    int idx = t + (i << 8);
    float v = (idx < KVdim) ? (row[idx] - m) * rstd : -3.0e38f;
    x[i] = v;
    lmax = fmaxf(lmax, v);
  }
  red[t] = lmax; __syncthreads();
  for (int k = 128; k > 0; k >>= 1) { if (t < k) red[t] = fmaxf(red[t], red[t + k]); __syncthreads(); }
  float bmax = red[0]; __syncthreads();

  float lsum = 0.f;
#pragma unroll
  for (int i = 0; i < 4; ++i) {
    int idx = t + (i << 8);
    if (idx < KVdim) { x[i] = __expf(x[i] - bmax); lsum += x[i]; }
  }
  red[t] = lsum; __syncthreads();
  for (int k = 128; k > 0; k >>= 1) { if (t < k) red[t] += red[t + k]; __syncthreads(); }
  float inv = 1.0f / red[0];
#pragma unroll
  for (int i = 0; i < 4; ++i) {
    int idx = t + (i << 8);
    if (idx < KVdim) orow[idx] = f2h(x[i] * inv);
  }
}

// Head mean + transpose: ctx [64][C][256] f32 -> ctxmean [16][256][C] f16
__global__ __launch_bounds__(256) void headmean_kernel(
    const float* __restrict__ ctx, unsigned short* __restrict__ outm,
    int Cdim, long long total)
{
  long long idx = (long long)blockIdx.x * 256 + threadIdx.x;
  if (idx >= total) return;
  int d = (int)(idx % Cdim);
  long long t = idx / Cdim;
  int n = (int)(t % 256);
  int b = (int)(t / 256);
  float s = 0.f;
#pragma unroll
  for (int h = 0; h < 4; ++h)
    s += ctx[((size_t)(b * 4 + h) * Cdim + d) * 256 + n];
  outm[idx] = f2h(0.25f * s);
}

// ---------------------------------------------------------------------------

extern "C" void kernel_launch(void* const* d_in, const int* in_sizes, int n_in,
                              void* d_out, int out_size, void* d_ws, size_t ws_size,
                              hipStream_t stream)
{
  (void)in_sizes; (void)n_in; (void)out_size; (void)ws_size;

  const float* embp[4] = {(const float*)d_in[0], (const float*)d_in[1],
                          (const float*)d_in[2], (const float*)d_in[3]};
  const float* emb_all = (const float*)d_in[4];
  const float* Wqp[4]  = {(const float*)d_in[5], (const float*)d_in[6],
                          (const float*)d_in[7], (const float*)d_in[8]};
  const float* Wk      = (const float*)d_in[9];
  const float* Wv      = (const float*)d_in[10];
  const float* Wop[4]  = {(const float*)d_in[11], (const float*)d_in[12],
                          (const float*)d_in[13], (const float*)d_in[14]};

  const int Bn = 16, Hn = 4, Nn = 196, NP = 256, KV = 960, Z = 64;
  const int Cs[4] = {64, 128, 256, 512};

  char* ws = (char*)d_ws;
  size_t off = 0;
  auto alloc = [&](size_t bytes) -> void* {
    void* p = ws + off;
    off += (bytes + 255) & ~(size_t)255;
    return p;
  };
  unsigned short* embAll_h = (unsigned short*)alloc((size_t)Bn * NP * KV * 2);
  unsigned short* Wk_h     = (unsigned short*)alloc((size_t)Hn * KV * KV * 2);
  unsigned short* Wv_h     = (unsigned short*)alloc((size_t)Hn * KV * KV * 2);
  unsigned short* Kt_h     = (unsigned short*)alloc((size_t)Z * KV * NP * 2);
  unsigned short* V_h      = (unsigned short*)alloc((size_t)Z * NP * KV * 2);
  unsigned short* emb_h    = (unsigned short*)alloc((size_t)Bn * NP * 512 * 2);
  unsigned short* Wq_h     = (unsigned short*)alloc((size_t)Hn * 512 * 512 * 2);
  unsigned short* Wo_h     = (unsigned short*)alloc((size_t)512 * 512 * 2);
  unsigned short* Qt_h     = (unsigned short*)alloc((size_t)Z * 512 * NP * 2);
  float*          scores   = (float*)alloc((size_t)Z * 512 * KV * 4);
  unsigned short* probs_h  = (unsigned short*)alloc((size_t)Z * 512 * KV * 2);
  float*          ctx      = (float*)alloc((size_t)Z * 512 * NP * 4);
  unsigned short* ctxm_h   = (unsigned short*)alloc((size_t)Bn * NP * 512 * 2);
  float*          stats    = (float*)alloc((size_t)Z * 2 * 4);

  auto cvt = [&](const float* in, unsigned short* out, int rows, int rowsPad,
                 int cols, int batch) {
    long long tot = (long long)batch * rowsPad * cols;
    int grid = (int)((tot + 255) / 256);
    convert_pad_kernel<<<dim3(grid), dim3(256), 0, stream>>>(in, out, rows, rowsPad, cols, tot);
  };
  auto gemm = [&](const unsigned short* A, const unsigned short* Bt, void* C,
                  int M, int N, int K, int lda, int ldb, int ldc,
                  long long aSB, long long aSH, long long bSB, long long bSH,
                  long long cSB, long long cSH,
                  int batch, int H, int cRowBound, float scale, int outF32) {
    dim3 grid(N / 64, M / 64, batch);
    wmma_gemm_kernel<<<grid, dim3(32), 0, stream>>>(A, Bt, C, K, lda, ldb, ldc,
                                                    aSB, aSH, bSB, bSH, cSB, cSH,
                                                    H, cRowBound, scale, outF32);
  };

  // --- shared K / V projections -------------------------------------------
  cvt(emb_all, embAll_h, Nn, NP, KV, Bn);
  cvt(Wk, Wk_h, KV, KV, KV, Hn);
  cvt(Wv, Wv_h, KV, KV, KV, Hn);

  // Kt[b,h][o,n] = sum_k Wk[h][o,k] * emb_all[b][n,k]   (stored K-transposed)
  gemm(Wk_h, embAll_h, Kt_h, KV, NP, KV, KV, KV, NP,
       0, (long long)KV * KV, (long long)NP * KV, 0,
       (long long)Hn * KV * NP, (long long)KV * NP, Z, Hn, KV, 1.0f, 0);
  // V[b,h][n,o] = sum_k emb_all[b][n,k] * Wv[h][o,k]
  gemm(embAll_h, Wv_h, V_h, NP, KV, KV, KV, KV, KV,
       (long long)NP * KV, 0, 0, (long long)KV * KV,
       (long long)Hn * NP * KV, (long long)NP * KV, Z, Hn, NP, 1.0f, 0);

  // --- four branches -------------------------------------------------------
  const float scl = 0.03227486121839514f; // 1/sqrt(960)
  size_t outOff = 0;
  for (int i = 0; i < 4; ++i) {
    int c = Cs[i];
    cvt(embp[i], emb_h, Nn, NP, c, Bn);
    cvt(Wqp[i], Wq_h, c, c, c, Hn);
    cvt(Wop[i], Wo_h, c, c, c, 1);

    // Qt[b,h][d,n] = sum_e Wq[h][d,e] * emb[b][n,e]
    gemm(Wq_h, emb_h, Qt_h, c, NP, c, c, c, NP,
         0, (long long)c * c, (long long)NP * c, 0,
         (long long)Hn * c * NP, (long long)c * NP, Z, Hn, c, 1.0f, 0);
    // scores[b,h][d,o] = scl * sum_n Qt[d,n] * Kt[o,n]
    gemm(Qt_h, Kt_h, scores, c, KV, NP, NP, NP, KV,
         (long long)Hn * c * NP, (long long)c * NP,
         (long long)Hn * KV * NP, (long long)KV * NP,
         (long long)Hn * c * KV, (long long)c * KV, Z, Hn, c, scl, 1);
    // instance-norm stats + softmax -> f16 probs
    stats_kernel<<<dim3(Z), dim3(256), 0, stream>>>(scores, stats, c * KV);
    norm_softmax_kernel<<<dim3(c, Z), dim3(256), 0, stream>>>(scores, probs_h, stats, c, KV);
    // ctx[b,h][d,n] = sum_o probs[d,o] * V[n,o]
    gemm(probs_h, V_h, ctx, c, NP, KV, KV, KV, NP,
         (long long)Hn * c * KV, (long long)c * KV,
         (long long)Hn * NP * KV, (long long)NP * KV,
         (long long)Hn * c * NP, (long long)c * NP, Z, Hn, c, 1.0f, 1);
    // mean over heads + transpose -> [b][n][d] f16
    {
      long long tot = (long long)Bn * NP * c;
      int grid = (int)((tot + 255) / 256);
      headmean_kernel<<<dim3(grid), dim3(256), 0, stream>>>(ctx, ctxm_h, c, tot);
    }
    // O[b][n][e] = sum_d ctxmean[b][n][d] * Wo[e][d]  (rows n < 196 -> d_out)
    gemm(ctxm_h, Wo_h, (float*)d_out + outOff, NP, c, c, c, c, c,
         (long long)NP * c, 0, 0, 0,
         (long long)Nn * c, 0, Bn, 1, Nn, 1.0f, 1);
    outOff += (size_t)Bn * Nn * c;
  }
}
